// Transformer_90787018703438
// MI455X (gfx1250) — matrix operand order repaired and checked by reading, not compiled, not run on hardware
//
#include <hip/hip_runtime.h>
#include <hip/hip_bf16.h>
#include <math.h>

// ---- problem constants (match reference) ----
#define CB 4
#define CS 1024
#define CV 1024
#define CD 512
#define CH 8
#define CF 2048
#define CHD 4096
#define CL 2
#define CEPS 1e-5f
#define INV_SQRT_DV 0.044194173824159216f  // 1/sqrt(512)

typedef __attribute__((ext_vector_type(16))) _Float16 v16h;
typedef __attribute__((ext_vector_type(8)))  _Float16 v8h;
typedef __attribute__((ext_vector_type(8)))  float    v8f;

// =====================================================================
// Generic strided batched GEMM, fp32 in/out, f16 WMMA compute.
//   C[m,n] = sum_k A[m*sAm + k] * B[k*sBk + n*sBn]  (+bias[m]) (relu) (+resid)
// A is always k-contiguous (true for every operand in this network).
// B is either n-contiguous (sBn==1) or k-contiguous (sBk==1).
// Block tile 64x64, BK=32, 256 threads = 8 waves (4 in M x 2 in N).
// LDS double-buffered; K-loop unrolled by 2 tiles so buffer indices are
// compile-time constants (no runtime ping-pong select in the hot loop).
// All K values here are multiples of 64.
// =====================================================================
__global__ __launch_bounds__(256) void gemm_wmma_f16(
    const float* __restrict__ A, long sAm, long bsA,
    const float* __restrict__ B, long sBk, long sBn, long bsB,
    float* __restrict__ C, long sCm, long sCn, long bsC,
    const float* __restrict__ bias,
    const float* __restrict__ resid, long bsR,
    int K, int relu)
{
    // padded row stride 40 halves (80B): 16B-aligned rows, conflict-free b128 access
    __shared__ _Float16 As[2][64 * 40];
    __shared__ _Float16 Bs[2][64 * 40];  // stored transposed: Bs[n][k]

    const int tid  = threadIdx.x;
    const int lane = tid & 31;
    const int wave = tid >> 5;
    const int wm   = wave >> 1;   // 0..3
    const int wn   = wave & 1;    // 0..1
    const int bm0  = blockIdx.x * 64;
    const int bn0  = blockIdx.y * 64;
    const int bz   = blockIdx.z;

    const float* Ab = A + (long)bz * bsA;
    const float* Bb = B + (long)bz * bsB;
    float*       Cb = C + (long)bz * bsC;

    // ---- per-thread staging geometry (hoisted out of the K loop) ----
    // A: thread covers row am, 8 consecutive k at ac
    const int am = tid >> 2, ac = (tid & 3) * 8;
    const float* pA = Ab + (long)(bm0 + am) * sAm + ac;
    const int aIdx = am * 40 + ac;                      // (80*am + 16*(tid&3)) B -> 16B aligned
    // B path NB (sBn==1): thread covers k-row nbK, 8 consecutive n at nbN
    // B path KB (sBk==1): thread covers n-col kbN, 8 consecutive k at kbK
    const bool contigN = (sBn == 1);
    const int nbK = tid >> 3, nbN = (tid & 7) * 8;
    const int kbN = tid >> 2, kbK = (tid & 3) * 8;
    const float* pB = contigN ? (Bb + (long)nbK * sBk + (bn0 + nbN))
                              : (Bb + (long)(bn0 + kbN) * sBn + kbK);

    float la[8], lb[8];
    auto loadRegs = [&](int k0) {
        const float4* qa = (const float4*)(pA + k0);
        float4 a0 = qa[0], a1 = qa[1];
        la[0]=a0.x; la[1]=a0.y; la[2]=a0.z; la[3]=a0.w;
        la[4]=a1.x; la[5]=a1.y; la[6]=a1.z; la[7]=a1.w;
        const float4* qb = contigN ? (const float4*)(pB + (long)k0 * sBk)
                                   : (const float4*)(pB + k0);
        float4 b0 = qb[0], b1 = qb[1];
        lb[0]=b0.x; lb[1]=b0.y; lb[2]=b0.z; lb[3]=b0.w;
        lb[4]=b1.x; lb[5]=b1.y; lb[6]=b1.z; lb[7]=b1.w;
    };
    auto storeLds = [&](int pb) {   // pb is always a literal -> immediate LDS offsets
        v8h ha;
        #pragma unroll
        for (int j = 0; j < 8; ++j) ha[j] = (_Float16)la[j];
        *(v8h*)&As[pb][aIdx] = ha;                       // one ds_store_b128
        if (contigN) {
            #pragma unroll
            for (int j = 0; j < 8; ++j)                  // transpose scatter
                Bs[pb][(nbN + j) * 40 + nbK] = (_Float16)lb[j];
        } else {
            v8h hb;
            #pragma unroll
            for (int j = 0; j < 8; ++j) hb[j] = (_Float16)lb[j];
            *(v8h*)&Bs[pb][kbN * 40 + kbK] = hb;         // one ds_store_b128
        }
    };

    // fragment geometry
    const int arow = wm * 16 + (lane & 15);
    const int ka   = (lane >> 4) * 16;
    const int kb   = (lane >> 4) * 16;
    const int n0f  = wn * 32 + (lane & 15);

    v8f acc0 = {}; v8f acc1 = {};
    auto mma = [&](int pb) {   // pb literal
        // A fragment (16x32 f16): lanes 0-15 hold K 0-15, lanes 16-31 K 16-31
        v8h alo = *(const v8h*)&As[pb][arow * 40 + ka];
        v8h ahi = *(const v8h*)&As[pb][arow * 40 + ka + 8];
        v16h a;
        #pragma unroll
        for (int j = 0; j < 8; ++j) { a[j] = alo[j]; a[8 + j] = ahi[j]; }
        // B fragments (32x16 f16): element j -> K = j + 16*(lane>=16), N = lane&15
        v8h b0lo = *(const v8h*)&Bs[pb][n0f * 40 + kb];
        v8h b0hi = *(const v8h*)&Bs[pb][n0f * 40 + kb + 8];
        v8h b1lo = *(const v8h*)&Bs[pb][(n0f + 16) * 40 + kb];
        v8h b1hi = *(const v8h*)&Bs[pb][(n0f + 16) * 40 + kb + 8];
        v16h b0, b1;
        #pragma unroll
        for (int j = 0; j < 8; ++j) { b0[j] = b0lo[j]; b0[8 + j] = b0hi[j];
                                      b1[j] = b1lo[j]; b1[8 + j] = b1hi[j]; }
        acc0 = __builtin_amdgcn_wmma_f32_16x16x32_f16(false, a, false, b0, (short)0, acc0, false, false);
        acc1 = __builtin_amdgcn_wmma_f32_16x16x32_f16(false, a, false, b1, (short)0, acc1, false, false);
    };

    loadRegs(0);
    storeLds(0);

    for (int k0 = 0; k0 < K; k0 += 64) {     // K is a multiple of 64
        __syncthreads();
        if (k0 + 64 < K) {                   // speculative prefetch one pair ahead
            __builtin_prefetch(pA + k0 + 64, 0, 1);
            __builtin_prefetch(contigN ? (pB + (long)(k0 + 64) * sBk) : (pB + k0 + 64), 0, 1);
        }
        loadRegs(k0 + 32);                   // always a valid tile (k0+32 <= K-32)
        mma(0);                              // compute from buf0 while VMEM in flight
        storeLds(1);                         // fill buf1 (disjoint from buf0 reads)
        __syncthreads();
        const bool more = (k0 + 64) < K;
        if (more) loadRegs(k0 + 64);
        mma(1);                              // compute from buf1
        if (more) storeLds(0);               // refill buf0 for next iteration
    }

    // epilogue: C element i of lane L -> row = i + 8*(L>=16), col = L&15
    const int rbase = bm0 + wm * 16 + (lane >> 4) * 8;
    const int c0 = bn0 + wn * 32 + (lane & 15);
    const int c1 = c0 + 16;
    const float* Rb = resid ? resid + (long)bz * bsR : nullptr;
    #pragma unroll
    for (int i = 0; i < 8; ++i) {
        const int r = rbase + i;
        float v0 = acc0[i], v1 = acc1[i];
        if (bias) { float bv = bias[r]; v0 += bv; v1 += bv; }
        if (relu) { v0 = fmaxf(v0, 0.f); v1 = fmaxf(v1, 0.f); }
        long o0 = (long)r * sCm + (long)c0 * sCn;
        long o1 = (long)r * sCm + (long)c1 * sCn;
        if (Rb) { v0 += Rb[o0]; v1 += Rb[o1]; }
        Cb[o0] = v0; Cb[o1] = v1;
    }
}

// x[b,d,s] += posenc(d,s)
__global__ __launch_bounds__(256) void posenc_add(float* __restrict__ x) {
    long idx = (long)blockIdx.x * 256 + threadIdx.x;   // over B*D*S
    int s = (int)(idx & (CS - 1));
    int d = (int)((idx >> 10) & (CD - 1));
    int i = d >> 1;
    float ang = (float)s * __expf(-9.210340371976184f * ((float)i / (float)CD));
    x[idx] += (d & 1) ? __cosf(ang) : __sinf(ang);
}

// faithful _mh reshape: dst[b][h][s][d] = raw_flat_per_batch[s*HD + h*D + d]
__global__ __launch_bounds__(256) void mh_split(const float* __restrict__ raw, float* __restrict__ dst) {
    long idx = (long)blockIdx.x * 256 + threadIdx.x;   // over B*H*S*D, ordered (b,h,s,d)
    int d = (int)(idx & (CD - 1));
    long r = idx >> 9;
    int s = (int)(r & (CS - 1));
    long r2 = r >> 10;
    int h = (int)(r2 & (CH - 1));
    int b = (int)(r2 >> 3);
    dst[idx] = raw[(long)b * CHD * CS + (long)s * CHD + (long)h * CD + d];
}

// softmax over the HEADS axis of scores[B,H,S,S], with optional causal zeroing (t<s)
__global__ __launch_bounds__(256) void softmax_heads(float* __restrict__ sc, int masked) {
    long idx = (long)blockIdx.x * 256 + threadIdx.x;   // over B*S*S, ordered (b,s,t)
    int t = (int)(idx & (CS - 1));
    long r = idx >> 10;
    int s = (int)(r & (CS - 1));
    int b = (int)(r >> 10);
    long base = (long)b * CH * CS * CS + (long)s * CS + t;
    float v[CH]; float mx = -3.4e38f;
    #pragma unroll
    for (int h = 0; h < CH; ++h) { v[h] = sc[base + (long)h * CS * CS] * INV_SQRT_DV; mx = fmaxf(mx, v[h]); }
    float sum = 0.f;
    #pragma unroll
    for (int h = 0; h < CH; ++h) { v[h] = __expf(v[h] - mx); sum += v[h]; }
    float inv = 1.f / sum;
    int kill = masked && (t < s);
    #pragma unroll
    for (int h = 0; h < CH; ++h) sc[base + (long)h * CS * CS] = kill ? 0.f : v[h] * inv;
}

// per-channel mean/var over (batch, seq) for BatchNorm1d (training math)
__global__ __launch_bounds__(256) void bn_stats_kernel(const float* __restrict__ att, float* __restrict__ mv) {
    const int d = blockIdx.x;
    __shared__ float s1[256], s2[256];
    float a = 0.f, q = 0.f;
    for (int i = threadIdx.x; i < CB * CS; i += 256) {
        int b = i >> 10, ss = i & (CS - 1);
        float v = att[((long)b * CD + d) * CS + ss];
        a += v; q += v * v;
    }
    s1[threadIdx.x] = a; s2[threadIdx.x] = q;
    __syncthreads();
    for (int off = 128; off > 0; off >>= 1) {
        if (threadIdx.x < off) { s1[threadIdx.x] += s1[threadIdx.x + off]; s2[threadIdx.x] += s2[threadIdx.x + off]; }
        __syncthreads();
    }
    if (threadIdx.x == 0) {
        float inv = 1.f / (float)(CB * CS);
        float m = s1[0] * inv;
        mv[d] = m;
        mv[CD + d] = s2[0] * inv - m * m;
    }
}

// x += gamma*(att-mean)*rsqrt(var+eps)+beta   (BN + residual, in place on stream)
__global__ __launch_bounds__(256) void bn_apply(const float* __restrict__ att, const float* __restrict__ mv,
                                                const float* __restrict__ g, const float* __restrict__ bb,
                                                float* __restrict__ x) {
    long idx = (long)blockIdx.x * 256 + threadIdx.x;   // over B*D*S
    int d = (int)((idx >> 10) & (CD - 1));
    float m = mv[d], v = mv[CD + d];
    x[idx] += g[d] * (att[idx] - m) * rsqrtf(v + CEPS) + bb[d];
}

// softmax over vocab axis: one block per (b,s) column, stride-S gathers
__global__ __launch_bounds__(256) void softmax_vocab(const float* __restrict__ logits, float* __restrict__ out) {
    int col = blockIdx.x;
    int b = col >> 10, s = col & (CS - 1);
    const float* p = logits + (long)b * CV * CS + s;
    float*       o = out    + (long)b * CV * CS + s;
    __shared__ float red[256];
    float mx = -3.4e38f;
    for (int v = threadIdx.x; v < CV; v += 256) mx = fmaxf(mx, p[(long)v * CS]);
    red[threadIdx.x] = mx; __syncthreads();
    for (int off = 128; off > 0; off >>= 1) {
        if (threadIdx.x < off) red[threadIdx.x] = fmaxf(red[threadIdx.x], red[threadIdx.x + off]);
        __syncthreads();
    }
    mx = red[0]; __syncthreads();
    float sum = 0.f;
    for (int v = threadIdx.x; v < CV; v += 256) sum += __expf(p[(long)v * CS] - mx);
    red[threadIdx.x] = sum; __syncthreads();
    for (int off = 128; off > 0; off >>= 1) {
        if (threadIdx.x < off) red[threadIdx.x] += red[threadIdx.x + off];
        __syncthreads();
    }
    float inv = 1.f / red[0];
    for (int v = threadIdx.x; v < CV; v += 256) o[(long)v * CS] = __expf(p[(long)v * CS] - mx) * inv;
}

// ---------------- host orchestration ----------------
struct AttnW { const float *wq,*bq,*wk,*bk,*wv,*bv,*wo,*bo,*g,*bb; };

static inline void gemm(hipStream_t st,
                        const float* A, long sAm, long bsA,
                        const float* B, long sBk, long sBn, long bsB,
                        float* C, long sCm, long sCn, long bsC,
                        const float* bias, const float* resid, long bsR,
                        int M, int N, int K, int nb, int relu) {
    dim3 grid(M / 64, N / 64, nb);
    gemm_wmma_f16<<<grid, dim3(256), 0, st>>>(A, sAm, bsA, B, sBk, sBn, bsB,
                                              C, sCm, sCn, bsC, bias, resid, bsR, K, relu);
}

extern "C" void kernel_launch(void* const* d_in, const int* in_sizes, int n_in,
                              void* d_out, int out_size, void* d_ws, size_t ws_size,
                              hipStream_t stream) {
    (void)in_sizes; (void)n_in; (void)out_size; (void)ws_size;
    const float* enc_tok = (const float*)d_in[0];
    const float* dec_tok = (const float*)d_in[1];
    const float* eemb_w  = (const float*)d_in[2];
    const float* eemb_b  = (const float*)d_in[3];
    AttnW ea { (const float*)d_in[4],(const float*)d_in[5],(const float*)d_in[6],(const float*)d_in[7],
               (const float*)d_in[8],(const float*)d_in[9],(const float*)d_in[10],(const float*)d_in[11],
               (const float*)d_in[12],(const float*)d_in[13] };
    const float* ef_w1=(const float*)d_in[14]; const float* ef_b1=(const float*)d_in[15];
    const float* ef_w2=(const float*)d_in[16]; const float* ef_b2=(const float*)d_in[17];
    const float* demb_w=(const float*)d_in[18]; const float* demb_b=(const float*)d_in[19];
    AttnW sa { (const float*)d_in[20],(const float*)d_in[21],(const float*)d_in[22],(const float*)d_in[23],
               (const float*)d_in[24],(const float*)d_in[25],(const float*)d_in[26],(const float*)d_in[27],
               (const float*)d_in[28],(const float*)d_in[29] };
    AttnW ca { (const float*)d_in[30],(const float*)d_in[31],(const float*)d_in[32],(const float*)d_in[33],
               (const float*)d_in[34],(const float*)d_in[35],(const float*)d_in[36],(const float*)d_in[37],
               (const float*)d_in[38],(const float*)d_in[39] };
    const float* df_w1=(const float*)d_in[40]; const float* df_b1=(const float*)d_in[41];
    const float* df_w2=(const float*)d_in[42]; const float* df_b2=(const float*)d_in[43];
    const float* out_w=(const float*)d_in[44]; const float* out_b=(const float*)d_in[45];

    // workspace layout (fp32), ~430 MB with reuse
    float* f = (float*)d_ws;
    const size_t nX   = (size_t)CB * CD * CS;        // 2,097,152
    const size_t nRaw = (size_t)CB * CHD * CS;       // 16,777,216
    const size_t nSc  = (size_t)CB * CH * CS * CS;   // 33,554,432
    float* xe  = f;              // encoder stream / enc_out
    float* xd  = xe + nX;        // decoder stream
    float* att = xd + nX;        // attention-block output (pre-BN)
    float* raw = att + nX;       // QKV raw out; reused as PV output and final logits
    float* qm  = raw + nRaw;     // [B,H,S,D]; reused as FFN hidden
    float* km  = qm + nRaw;
    float* vm  = km + nRaw;
    float* sc  = vm + nRaw;      // [B,H,S,S] scores/probs
    float* mv  = sc + nSc;       // [2*D] BN mean/var
    float* obuf = raw;
    float* hid  = qm;
    float* logits = raw;

    const long bsX = (long)CD * CS, bsRaw = (long)CHD * CS;
    const long bsQ = (long)CS * CD, bsS = (long)CS * CS;
    const int nSplitBlk = (int)(nRaw / 256);

    auto attn = [&](const float* kin, const float* qin, const float* vin,
                    float* x, const AttnW& w, int l, int masked) {
        const float *wq=w.wq+(long)l*CHD*CD, *bq=w.bq+(long)l*CHD;
        const float *wk=w.wk+(long)l*CHD*CD, *bk=w.bk+(long)l*CHD;
        const float *wv=w.wv+(long)l*CHD*CD, *bv=w.bv+(long)l*CHD;
        const float *wo=w.wo+(long)l*CD*CHD, *bo=w.bo+(long)l*CD;
        const float *g =w.g +(long)l*CD,     *bb=w.bb+(long)l*CD;
        // QKV projections (W[HD,D] @ X[D,S] per batch) + faithful _mh permutation
        gemm(stream, wq, CD,0, qin, CS,1,bsX, raw, CS,1,bsRaw, bq, nullptr,0, CHD, CS, CD, CB, 0);
        mh_split<<<nSplitBlk,256,0,stream>>>(raw, qm);
        gemm(stream, wk, CD,0, kin, CS,1,bsX, raw, CS,1,bsRaw, bk, nullptr,0, CHD, CS, CD, CB, 0);
        mh_split<<<nSplitBlk,256,0,stream>>>(raw, km);
        gemm(stream, wv, CD,0, vin, CS,1,bsX, raw, CS,1,bsRaw, bv, nullptr,0, CHD, CS, CD, CB, 0);
        mh_split<<<nSplitBlk,256,0,stream>>>(raw, vm);
        // scores[b,h,s,t] = Q[s,:] . K[t,:]  (batched over B*H; B operand k-contiguous)
        gemm(stream, qm, CD,bsQ, km, 1,CD,bsQ, sc, CS,1,bsS, nullptr,nullptr,0, CS, CS, CD, CB*CH, 0);
        softmax_heads<<<(int)((size_t)CB*CS*CS/256),256,0,stream>>>(sc, masked);
        // O[b][h*D+d][s] = sum_t P[s,t] * V[t,d]   (strided C store -> [HD,S] concat layout)
        gemm(stream, sc, CS,bsS, vm, CD,1,bsQ, obuf, 1,CS,(long)CD*CS, nullptr,nullptr,0, CS, CD, CS, CB*CH, 0);
        // out projection Wo[D,HD] @ O[HD,S]
        gemm(stream, wo, CHD,0, obuf, CS,1,bsRaw, att, CS,1,bsX, bo, nullptr,0, CD, CS, CHD, CB, 0);
        // BN over (batch, seq) + residual into stream
        bn_stats_kernel<<<CD,256,0,stream>>>(att, mv);
        bn_apply<<<(int)(nX/256),256,0,stream>>>(att, mv, g, bb, x);
    };

    auto ffn = [&](float* x, const float* w1,const float* b1,const float* w2,const float* b2,int l) {
        gemm(stream, w1+(long)l*CF*CD, CD,0, x, CS,1,bsX, hid, CS,1,(long)CF*CS,
             b1+(long)l*CF, nullptr,0, CF, CS, CD, CB, 1);
        gemm(stream, w2+(long)l*CD*CF, CF,0, hid, CS,1,(long)CF*CS, x, CS,1,bsX,
             b2+(long)l*CD, x, bsX, CD, CS, CF, CB, 0);
    };

    // ---- Encoder ----
    gemm(stream, eemb_w, CV,0, enc_tok, CS,1,(long)CV*CS, xe, CS,1,bsX, eemb_b, nullptr,0, CD, CS, CV, CB, 0);
    posenc_add<<<(int)(nX/256),256,0,stream>>>(xe);
    for (int l = 0; l < CL; ++l) {
        attn(xe, xe, xe, xe, ea, l, 0);
        ffn(xe, ef_w1, ef_b1, ef_w2, ef_b2, l);
    }
    // xe now holds enc_out

    // ---- Decoder ----
    gemm(stream, demb_w, CV,0, dec_tok, CS,1,(long)CV*CS, xd, CS,1,bsX, demb_b, nullptr,0, CD, CS, CV, CB, 0);
    posenc_add<<<(int)(nX/256),256,0,stream>>>(xd);
    for (int l = 0; l < CL; ++l) {
        attn(xd, xd, xd, xd, sa, l, 1);          // masked self-attention
        attn(xe, xe, xd, xd, ca, l, 0);          // cross: K,Q from enc_out, V from dec stream
        ffn(xd, df_w1, df_b1, df_w2, df_b2, l);
    }

    // ---- Final projection + vocab softmax ----
    gemm(stream, out_w, CD,0, xd, CS,1,bsX, logits, CS,1,(long)CV*CS, out_b, nullptr,0, CV, CS, CD, CB, 0);
    softmax_vocab<<<CB*CS,256,0,stream>>>(logits, (float*)d_out);
}
